// GENERATOR_70712341562227
// MI455X (gfx1250) — matrix operand order repaired and checked by reading, not compiled, not run on hardware
//
#include <hip/hip_runtime.h>
#include <stdint.h>

// Sizes from the reference
#define N_VOC 32000
#define EMB   512
#define HID   512
#define H3    1536
#define NOISE 100
#define IMG   512
#define SEQ   32
#define BATCH 64
#define ROWS  (SEQ * BATCH)          // 2048 rows of hidden states for output GEMM
#define NT    4                      // n-tiles (of 16) per wave in the big GEMM
#define NCHUNK (N_VOC / (16 * NT))   // 500 partial-softmax chunks per row

typedef __attribute__((ext_vector_type(16))) __bf16 v16bf;
typedef __attribute__((ext_vector_type(8)))  float  v8f;

union ABfrag { v16bf v; uint32_t u[8]; };

__device__ __forceinline__ unsigned short f2bf(float f) {
    uint32_t u = __float_as_uint(f);
    uint32_t r = (u + 0x7FFFu + ((u >> 16) & 1u)) >> 16;
    return (unsigned short)r;
}
__device__ __forceinline__ float bf2f(unsigned short h) {
    return __uint_as_float(((uint32_t)h) << 16);
}
__device__ __forceinline__ void load8(const unsigned short* p, uint32_t* dst) {
    const uint4* q = reinterpret_cast<const uint4*>(p);
    uint4 t = *q;
    dst[0] = t.x; dst[1] = t.y; dst[2] = t.z; dst[3] = t.w;
}
// A fragment: 16x32 bf16, A row-major [16, lda], lane m = lane&15
__device__ __forceinline__ v16bf load_a_frag(const unsigned short* A, int lda, int k0, int lane) {
    int m = lane & 15, half = lane >> 4;
    const unsigned short* r = A + m * lda;
    ABfrag f;
    load8(r + k0 + (half ? 8 : 0),      f.u);
    load8(r + k0 + 16 + (half ? 8 : 0), f.u + 4);
    return f.v;
}
// B fragment: 32x16 bf16 where B[k][n] = W[n][k], W row-major [16 rows starting at W, ldw]
__device__ __forceinline__ v16bf load_b_frag(const unsigned short* W, int ldw, int k0, int lane) {
    int n = lane & 15, half = lane >> 4;
    const unsigned short* r = W + n * ldw + k0 + (half ? 8 : 0);
    ABfrag f;
    load8(r,      f.u);
    load8(r + 16, f.u + 4);
    return f.v;
}
__device__ __forceinline__ float sigmoidf(float x) { return 1.0f / (1.0f + __expf(-x)); }

// ---------------- small utility kernels ----------------
__global__ void k_cvt_bf16(const float* __restrict__ src, unsigned short* __restrict__ dst, int n) {
    int i = blockIdx.x * blockDim.x + threadIdx.x;
    if (i < n) dst[i] = f2bf(src[i]);
}
// X_bf[t][b][k] = bf16(emb[inp[b][t]][k])
__global__ void k_gather_x(const int* __restrict__ inp, const float* __restrict__ emb,
                           unsigned short* __restrict__ Xbf) {
    int idx = blockIdx.x * blockDim.x + threadIdx.x;  // < SEQ*BATCH*EMB
    if (idx >= SEQ * BATCH * EMB) return;
    int k = idx & 511, b = (idx >> 9) & 63, t = idx >> 15;
    int tok = inp[b * SEQ + t];
    Xbf[idx] = f2bf(emb[tok * EMB + k]);
}
__global__ void k_zero_h(float* __restrict__ h, unsigned short* __restrict__ hbf) {
    int i = blockIdx.x * blockDim.x + threadIdx.x;
    if (i < BATCH * HID) { h[i] = 0.0f; hbf[i] = 0; }
}
// embed_img = img @ W_enc.T + b_enc  (64x4096 @ 4096x512, 0.3 GFLOP — VALU is fine)
__global__ void k_enc(const float* __restrict__ img, const float* __restrict__ Wenc,
                      const float* __restrict__ benc, float* __restrict__ eimg) {
    int idx = blockIdx.x * blockDim.x + threadIdx.x;  // 64*512
    if (idx >= BATCH * IMG) return;
    int b = idx >> 9, i = idx & 511;
    const float4* x = reinterpret_cast<const float4*>(img + b * 4096);
    const float4* w = reinterpret_cast<const float4*>(Wenc + i * 4096);
    float acc = 0.f;
    for (int k = 0; k < 1024; ++k) {
        float4 xv = x[k], wv = w[k];
        acc += xv.x * wv.x + xv.y * wv.y + xv.z * wv.z + xv.w * wv.w;
    }
    eimg[idx] = acc + benc[i];
}
// x0 = relu(concat(eimg, noise) @ W_e2i.T + b_e2i)  -> bf16
__global__ void k_e2i(const float* __restrict__ eimg, const float* __restrict__ noise,
                      const float* __restrict__ We2i, const float* __restrict__ be2i,
                      unsigned short* __restrict__ x0bf) {
    int idx = blockIdx.x * blockDim.x + threadIdx.x;  // 64*512
    if (idx >= BATCH * EMB) return;
    int b = idx >> 9, e = idx & 511;
    const float* w = We2i + e * (IMG + NOISE);
    float acc = be2i[e];
    const float* ev = eimg + b * IMG;
    for (int k = 0; k < IMG; ++k) acc += ev[k] * w[k];
    const float* nv = noise + b * NOISE;
    for (int k = 0; k < NOISE; ++k) acc += nv[k] * w[IMG + k];
    x0bf[idx] = f2bf(fmaxf(acc, 0.0f));
}

// ---------------- GRU step: two [64,512]x[512,1536] GEMMs via WMMA ----------------
__global__ __launch_bounds__(32)
void k_gru_gemm(const unsigned short* __restrict__ Ax, const unsigned short* __restrict__ Ah,
                const unsigned short* __restrict__ Wih, const unsigned short* __restrict__ Whh,
                const float* __restrict__ bih, const float* __restrict__ bhh,
                float* __restrict__ gi, float* __restrict__ gh) {
    int lane = threadIdx.x;
    int mt = blockIdx.x, nt = blockIdx.y, z = blockIdx.z;
    const unsigned short* A = (z ? Ah : Ax) + mt * 16 * HID;
    const unsigned short* W = (z ? Whh : Wih) + nt * 16 * HID;
    const float* bias = z ? bhh : bih;
    float* G = z ? gh : gi;
    v8f acc = {0.f,0.f,0.f,0.f,0.f,0.f,0.f,0.f};
    for (int kc = 0; kc < HID; kc += 32) {
        v16bf a = load_a_frag(A, HID, kc, lane);
        v16bf b = load_b_frag(W, HID, kc, lane);
        acc = __builtin_amdgcn_wmma_f32_16x16x32_bf16(false, a, false, b, (short)0, acc, false, false);
    }
    int col = nt * 16 + (lane & 15), half = lane >> 4;
    float bn = bias[col];
#pragma unroll
    for (int v = 0; v < 8; ++v) {
        int row = mt * 16 + v + 8 * half;
        G[row * H3 + col] = acc[v] + bn;
    }
}
// Elementwise GRU gates + h update; step>0 also records h into h_all (bf16 + f32)
__global__ void k_gate(const float* __restrict__ gi, const float* __restrict__ gh,
                       float* __restrict__ h, unsigned short* __restrict__ hbf,
                       unsigned short* __restrict__ hall_bf, float* __restrict__ hall_f,
                       int step) {
    int idx = blockIdx.x * blockDim.x + threadIdx.x;  // 64*512
    if (idx >= BATCH * HID) return;
    int b = idx >> 9, j = idx & 511;
    const float* gib = gi + b * H3;
    const float* ghb = gh + b * H3;
    float r = sigmoidf(gib[j] + ghb[j]);
    float z = sigmoidf(gib[HID + j] + ghb[HID + j]);
    float n = tanhf(gib[2 * HID + j] + r * ghb[2 * HID + j]);
    float hn = (1.0f - z) * n + z * h[idx];
    h[idx] = hn;
    unsigned short hb = f2bf(hn);
    hbf[idx] = hb;
    if (step > 0) {
        hall_bf[(step - 1) * (BATCH * HID) + idx] = hb;
        hall_f [(step - 1) * (BATCH * HID) + idx] = hn;
    }
}

// ---------------- big GEMM [2048,512]x[512,32000] + fused online softmax partials ----------------
__global__ __launch_bounds__(32)
void k_logits(const unsigned short* __restrict__ Hall, const unsigned short* __restrict__ Wout,
              const float* __restrict__ bout,
              float* __restrict__ pmax, float* __restrict__ psum) {
    int lane = threadIdx.x;
    int mt = blockIdx.x;   // 0..127 row tiles
    int ck = blockIdx.y;   // 0..NCHUNK-1 column chunks (NT n-tiles each)
    const unsigned short* A = Hall + mt * 16 * HID;
    v8f acc[NT];
#pragma unroll
    for (int j = 0; j < NT; ++j) acc[j] = (v8f){0.f,0.f,0.f,0.f,0.f,0.f,0.f,0.f};
    for (int kc = 0; kc < HID; kc += 32) {
        v16bf a = load_a_frag(A, HID, kc, lane);
#pragma unroll
        for (int j = 0; j < NT; ++j) {
            const unsigned short* W = Wout + (size_t)(ck * (16 * NT) + j * 16) * HID;
            v16bf b = load_b_frag(W, HID, kc, lane);
            acc[j] = __builtin_amdgcn_wmma_f32_16x16x32_bf16(false, a, false, b, (short)0, acc[j], false, false);
        }
    }
    int half = lane >> 4, nlo = lane & 15;
    float m[8], s[8];
#pragma unroll
    for (int v = 0; v < 8; ++v) { m[v] = -INFINITY; s[v] = 0.f; }
#pragma unroll
    for (int j = 0; j < NT; ++j) {
        float bn = bout[ck * (16 * NT) + j * 16 + nlo];
#pragma unroll
        for (int v = 0; v < 8; ++v) {
            float val = acc[j][v] + bn;
            // butterfly over the 16 columns (low 4 lane bits) — halves stay independent
            float tm = val;
            tm = fmaxf(tm, __shfl_xor(tm, 1));
            tm = fmaxf(tm, __shfl_xor(tm, 2));
            tm = fmaxf(tm, __shfl_xor(tm, 4));
            tm = fmaxf(tm, __shfl_xor(tm, 8));
            float e = __expf(val - tm);
            e += __shfl_xor(e, 1);
            e += __shfl_xor(e, 2);
            e += __shfl_xor(e, 4);
            e += __shfl_xor(e, 8);
            float nm = fmaxf(m[v], tm);
            s[v] = s[v] * __expf(m[v] - nm) + e * __expf(tm - nm);
            m[v] = nm;
        }
    }
    if (nlo == 0) {
#pragma unroll
        for (int v = 0; v < 8; ++v) {
            int row = mt * 16 + v + 8 * half;
            pmax[(size_t)ck * ROWS + row] = m[v];
            psum[(size_t)ck * ROWS + row] = s[v];
        }
    }
}
// combine NCHUNK partials per row -> logsumexp
__global__ void k_lse(const float* __restrict__ pmax, const float* __restrict__ psum,
                      float* __restrict__ lse) {
    int r = blockIdx.x * blockDim.x + threadIdx.x;
    if (r >= ROWS) return;
    float m = -INFINITY;
    for (int c = 0; c < NCHUNK; ++c) m = fmaxf(m, pmax[(size_t)c * ROWS + r]);
    float s = 0.f;
    for (int c = 0; c < NCHUNK; ++c) s += psum[(size_t)c * ROWS + r] * __expf(pmax[(size_t)c * ROWS + r] - m);
    lse[r] = m + __logf(s);
}
// target logit per row (same bf16 operands as the GEMM for consistency)
__global__ void k_tgt(const unsigned short* __restrict__ Hall, const unsigned short* __restrict__ Wout,
                      const float* __restrict__ bout, const int* __restrict__ target,
                      float* __restrict__ tl) {
    int r = blockIdx.x * blockDim.x + threadIdx.x;
    if (r >= ROWS) return;
    int t = r >> 6, b = r & 63;
    int tgt = target[b * SEQ + t];
    const unsigned short* hrow = Hall + (size_t)r * HID;
    const unsigned short* wrow = Wout + (size_t)tgt * HID;
    float acc = 0.f;
    for (int k = 0; k < HID; ++k) acc += bf2f(hrow[k]) * bf2f(wrow[k]);
    tl[r] = acc + bout[tgt];
}
__global__ void k_final(const float* __restrict__ lse, const float* __restrict__ tl,
                        float* __restrict__ out) {
    __shared__ float red[256];
    int tid = threadIdx.x;
    float a = 0.f;
    for (int r = tid; r < ROWS; r += 256) a += lse[r] - tl[r];
    red[tid] = a; __syncthreads();
    for (int st = 128; st > 0; st >>= 1) { if (tid < st) red[tid] += red[tid + st]; __syncthreads(); }
    if (tid == 0) out[0] = red[0] * (1.0f / (float)BATCH);
}

// ---------------- host ----------------
static inline size_t alignup(size_t x) { return (x + 255) & ~(size_t)255; }

extern "C" void kernel_launch(void* const* d_in, const int* in_sizes, int n_in,
                              void* d_out, int out_size, void* d_ws, size_t ws_size,
                              hipStream_t stream) {
    const int*   inp    = (const int*)  d_in[0];
    const int*   target = (const int*)  d_in[1];
    const float* img    = (const float*)d_in[2];
    const float* noise  = (const float*)d_in[3];
    const float* W_enc  = (const float*)d_in[4];
    const float* b_enc  = (const float*)d_in[5];
    const float* W_e2i  = (const float*)d_in[6];
    const float* b_e2i  = (const float*)d_in[7];
    const float* emb    = (const float*)d_in[8];
    const float* w_ih   = (const float*)d_in[9];
    const float* w_hh   = (const float*)d_in[10];
    const float* b_ih   = (const float*)d_in[11];
    const float* b_hh   = (const float*)d_in[12];
    const float* W_out  = (const float*)d_in[13];
    const float* b_out  = (const float*)d_in[14];
    float* out = (float*)d_out;

    char* ws = (char*)d_ws;
    size_t off = 0;
    auto take = [&](size_t bytes) { char* p = ws + off; off += alignup(bytes); return p; };
    unsigned short* Wout_bf = (unsigned short*)take((size_t)N_VOC * HID * 2);
    unsigned short* wih_bf  = (unsigned short*)take((size_t)H3 * HID * 2);
    unsigned short* whh_bf  = (unsigned short*)take((size_t)H3 * HID * 2);
    unsigned short* X_bf    = (unsigned short*)take((size_t)SEQ * BATCH * EMB * 2);
    unsigned short* x0_bf   = (unsigned short*)take((size_t)BATCH * EMB * 2);
    float*          h_f32   = (float*)take((size_t)BATCH * HID * 4);
    unsigned short* h_bf    = (unsigned short*)take((size_t)BATCH * HID * 2);
    float*          gi_ws   = (float*)take((size_t)BATCH * H3 * 4);
    float*          gh_ws   = (float*)take((size_t)BATCH * H3 * 4);
    unsigned short* hall_bf = (unsigned short*)take((size_t)ROWS * HID * 2);
    float*          hall_f  = (float*)take((size_t)ROWS * HID * 4);
    float*          eimg    = (float*)take((size_t)BATCH * IMG * 4);
    float*          pmax    = (float*)take((size_t)NCHUNK * ROWS * 4);
    float*          psum    = (float*)take((size_t)NCHUNK * ROWS * 4);
    float*          lse     = (float*)take((size_t)ROWS * 4);
    float*          tlog    = (float*)take((size_t)ROWS * 4);

    // convert weights to bf16 (W_out stays L2-resident at 32.8 MB)
    k_cvt_bf16<<<(N_VOC * HID + 255) / 256, 256, 0, stream>>>(W_out, Wout_bf, N_VOC * HID);
    k_cvt_bf16<<<(H3 * HID + 255) / 256, 256, 0, stream>>>(w_ih, wih_bf, H3 * HID);
    k_cvt_bf16<<<(H3 * HID + 255) / 256, 256, 0, stream>>>(w_hh, whh_bf, H3 * HID);
    // gather token embeddings for all timesteps
    k_gather_x<<<(SEQ * BATCH * EMB + 255) / 256, 256, 0, stream>>>(inp, emb, X_bf);
    // conditioning path
    k_enc<<<(BATCH * IMG + 255) / 256, 256, 0, stream>>>(img, W_enc, b_enc, eimg);
    k_e2i<<<(BATCH * EMB + 255) / 256, 256, 0, stream>>>(eimg, noise, W_e2i, b_e2i, x0_bf);
    k_zero_h<<<(BATCH * HID + 255) / 256, 256, 0, stream>>>(h_f32, h_bf);

    // 33 sequential GRU steps (step 0 = conditioning, steps 1..32 consume tokens)
    for (int s = 0; s <= SEQ; ++s) {
        const unsigned short* Ax = (s == 0) ? x0_bf : (X_bf + (size_t)(s - 1) * BATCH * EMB);
        k_gru_gemm<<<dim3(BATCH / 16, H3 / 16, 2), 32, 0, stream>>>(
            Ax, h_bf, wih_bf, whh_bf, b_ih, b_hh, gi_ws, gh_ws);
        k_gate<<<(BATCH * HID + 255) / 256, 256, 0, stream>>>(
            gi_ws, gh_ws, h_f32, h_bf, hall_bf, hall_f, s);
    }

    // dominant GEMM + fused online softmax partials
    k_logits<<<dim3(ROWS / 16, NCHUNK), 32, 0, stream>>>(hall_bf, Wout_bf, b_out, pmax, psum);
    k_lse<<<(ROWS + 255) / 256, 256, 0, stream>>>(pmax, psum, lse);
    k_tgt<<<(ROWS + 255) / 256, 256, 0, stream>>>(hall_bf, Wout_bf, b_out, target, tlog);
    k_final<<<1, 256, 0, stream>>>(lse, tlog, out);
}
